// FINNv1_binary_7971459301398
// MI455X (gfx1250) — 8-bit matrix operand order repaired and checked by reading, compiled, not run on hardware
//
#include <hip/hip_runtime.h>
#include <stdint.h>


typedef int v2i_t __attribute__((ext_vector_type(2)));
typedef int v4i_t __attribute__((ext_vector_type(4)));
typedef int v8i_t __attribute__((ext_vector_type(8)));

union Frag8 { v8i_t v; v2i_t h[4]; v4i_t q[2]; };

__global__ void k_pack_conv(const float* __restrict__ w, int8_t* __restrict__ wp,
                            int Co, int Ci, int KH, int KW) {
  int idx = blockIdx.x * blockDim.x + threadIdx.x;
  int total = Co * Ci * KH * KW;
  if (idx >= total) return;
  int ci = idx % Ci; int t = idx / Ci;
  int kx = t % KW;   t /= KW;
  int ky = t % KH;   int co = t / KH;
  float v = w[((co * Ci + ci) * KH + ky) * KW + kx];
  wp[idx] = (v >= 0.0f) ? (int8_t)1 : (int8_t)(-1);
}

__global__ void k_pack_fc(const float* __restrict__ w, int8_t* __restrict__ wp,
                          int Fo, int Fi, int Npad) {
  int idx = blockIdx.x * blockDim.x + threadIdx.x;
  int total = Npad * Fi;
  if (idx >= total) return;
  int row = idx / Fi, k = idx % Fi;
  wp[idx] = (row < Fo) ? ((w[row * Fi + k] >= 0.0f) ? (int8_t)1 : (int8_t)(-1))
                       : (int8_t)0;
}

__global__ void k_binarize_input(const float* __restrict__ x, int8_t* __restrict__ xb) {
  int idx = blockIdx.x * blockDim.x + threadIdx.x;
  const int total = 512 * 32 * 32 * 3;
  if (idx >= total) return;
  int c = idx % 3; int t = idx / 3;
  int w = t % 32;  t /= 32;
  int h = t % 32;  int n = t / 32;
  float v = x[((n * 3 + c) * 32 + h) * 32 + w];
  xb[idx] = (v >= 0.5f) ? (int8_t)1 : (int8_t)(-1);
}

__global__ void __launch_bounds__(256) k_conv0(
    const int8_t* __restrict__ xb, const int8_t* __restrict__ wp,
    const float* __restrict__ g, const float* __restrict__ b,
    const float* __restrict__ m, const float* __restrict__ v,
    int8_t* __restrict__ out) {
  __shared__ int8_t ws[64 * 27];
  for (int t = threadIdx.x; t < 64 * 27; t += 256) ws[t] = wp[t];
  __syncthreads();

  int idx = blockIdx.x * 256 + threadIdx.x;
  if (idx >= 512 * 30 * 30) return;
  int n = idx / 900; int r = idx % 900;
  int oy = r / 30, ox = r % 30;

  int win[27];
  #pragma unroll
  for (int ky = 0; ky < 3; ++ky) {
    int base = ((n * 32 + oy + ky) * 32 + ox) * 3;
    #pragma unroll
    for (int j = 0; j < 9; ++j) win[ky * 9 + j] = xb[base + j];
  }
  for (int co = 0; co < 64; ++co) {
    int s = 0;
    #pragma unroll
    for (int j = 0; j < 27; ++j) s += win[j] * (int)ws[co * 27 + j];
    float sc = g[co] * rsqrtf(v[co] + 1e-4f);
    float y = ((float)s - m[co]) * sc + b[co];
    out[(size_t)idx * 64 + co] = (y >= 0.0f) ? (int8_t)1 : (int8_t)(-1);
  }
}

__global__ void k_maxpool2x2(const int8_t* __restrict__ in, int8_t* __restrict__ out,
                             int H, int W, int C, int total) {
  int idx = blockIdx.x * blockDim.x + threadIdx.x;
  if (idx >= total) return;
  int c = idx % C; int t = idx / C;
  int ox = t % (W / 2); t /= (W / 2);
  int oy = t % (H / 2); int n = t / (H / 2);
  size_t base = (((size_t)n * H + 2 * oy) * W + 2 * ox) * C + c;
  int8_t a0 = in[base], a1 = in[base + C];
  int8_t a2 = in[base + (size_t)W * C], a3 = in[base + (size_t)W * C + C];
  int8_t mx = a0 > a1 ? a0 : a1;
  int8_t my = a2 > a3 ? a2 : a3;
  out[idx] = mx > my ? mx : my;
}

__global__ void __launch_bounds__(256) bnn_gemm_wmma(
    const int8_t* __restrict__ A, const int8_t* __restrict__ W,
    int8_t* __restrict__ out8, float* __restrict__ outf,
    const float* __restrict__ bn_g, const float* __restrict__ bn_b,
    const float* __restrict__ bn_m, const float* __restrict__ bn_v,
    int IH, int IW, int Cin, int OH, int OW, int Cout, int CoutTrue,
    int KH, int KW, float eps, int mode) {
  __shared__ alignas(16) int8_t Bt[2][64 * 64];

  const int lane = threadIdx.x & 31;
  const int wave = threadIdx.x >> 5;
  const int col16 = lane & 15;
  const int halfsel = (lane >> 4) & 1;
  const int Mbase = blockIdx.x * 256;
  const int colbase = blockIdx.y * 64;
  const int Ktot = KH * KW * Cin;
  const int ncb = Cin >> 6;
  const int nk = KH * KW * ncb;

  const int8_t* abase[2];
  #pragma unroll
  for (int ms = 0; ms < 2; ++ms) {
    int t = Mbase + wave * 32 + ms * 16 + col16;
    const int ox = t % OW; t /= OW;
    const int oy = t % OH; const int n = t / OH;
    abase[ms] = A + (((size_t)n * IH + oy) * IW + ox) * Cin + halfsel * 16;
  }

  const int bcol = threadIdx.x >> 2;
  const int bseg = threadIdx.x & 3;
  const int8_t* wrow = W + (size_t)(colbase + bcol) * Ktot + bseg * 16;
  const int bofs = bcol * 64 + bseg * 16;

  v8i_t acc[2][4];
  #pragma unroll
  for (int ms = 0; ms < 2; ++ms)
    #pragma unroll
    for (int nt = 0; nt < 4; ++nt) acc[ms][nt] = (v8i_t)0;

  *(v4i_t*)&Bt[0][bofs] = *(const v4i_t*)(wrow);
  __syncthreads();

  for (int ks = 0; ks < nk; ++ks) {
    const int cur = ks & 1;
    const bool haveNext = (ks + 1 < nk);

    v4i_t nextW;
    if (haveNext) {
      nextW = *(const v4i_t*)(wrow + (size_t)(ks + 1) * 64);
      if (ks + 2 < nk)
        __builtin_prefetch(wrow + (size_t)(ks + 2) * 64, 0, 0);
    }

    const int cb = ks % ncb;
    const int kk = ks / ncb;
    const int aoff = ((kk / KW) * IW + (kk % KW)) * Cin + cb * 64;

    Frag8 af[2];
    #pragma unroll
    for (int ms = 0; ms < 2; ++ms) {
      const int8_t* ap = abase[ms] + aoff;
      af[ms].h[0] = *(const v2i_t*)(ap);
      af[ms].h[1] = *(const v2i_t*)(ap + 8);
      af[ms].h[2] = *(const v2i_t*)(ap + 32);
      af[ms].h[3] = *(const v2i_t*)(ap + 40);
    }

    Frag8 bf[4];
    #pragma unroll
    for (int nt = 0; nt < 4; ++nt) {
      const int8_t* bp = &Bt[cur][(nt * 16 + col16) * 64 + halfsel * 16];
      bf[nt].q[0] = *(const v4i_t*)(bp);
      bf[nt].q[1] = *(const v4i_t*)(bp + 32);
    }

    #pragma unroll
    for (int ms = 0; ms < 2; ++ms)
      #pragma unroll
      for (int nt = 0; nt < 4; ++nt)
        acc[ms][nt] = __builtin_amdgcn_wmma_i32_16x16x64_iu8(
            true, af[ms].v, true, bf[nt].v, acc[ms][nt], false, false);

    if (haveNext) *(v4i_t*)&Bt[cur ^ 1][bofs] = nextW;
    __syncthreads();
  }

  if (mode == 0) {
    #pragma unroll
    for (int nt = 0; nt < 4; ++nt) {
      const int ch = colbase + nt * 16 + col16;
      const float sc = bn_g[ch] * rsqrtf(bn_v[ch] + eps);
      const float mm = bn_m[ch], bb = bn_b[ch];
      #pragma unroll
      for (int ms = 0; ms < 2; ++ms) {
        const int mrow0 = Mbase + wave * 32 + ms * 16 + halfsel * 8;
        #pragma unroll
        for (int r = 0; r < 8; ++r) {
          float y = ((float)acc[ms][nt][r] - mm) * sc + bb;
          out8[(size_t)(mrow0 + r) * Cout + ch] = (y >= 0.0f) ? (int8_t)1 : (int8_t)(-1);
        }
      }
    }
  } else {
    const float sc = bn_g[0] * rsqrtf(bn_v[0] + eps);
    const float mm = bn_m[0], bb = bn_b[0];
    #pragma unroll
    for (int nt = 0; nt < 4; ++nt) {
      const int ch = colbase + nt * 16 + col16;
      if (ch < CoutTrue) {
        #pragma unroll
        for (int ms = 0; ms < 2; ++ms) {
          const int mrow0 = Mbase + wave * 32 + ms * 16 + halfsel * 8;
          #pragma unroll
          for (int r = 0; r < 8; ++r) {
            float y = ((float)acc[ms][nt][r] - mm) * sc + bb;
            outf[(size_t)(mrow0 + r) * CoutTrue + ch] = y;
          }
        }
      }
    }
  }
}

static inline int ceil_div(int a, int b) { return (a + b - 1) / b; }

extern "C" void kernel_launch(void* const* d_in, const int* in_sizes, int n_in,
                              void* d_out, int out_size, void* d_ws, size_t ws_size,
                              hipStream_t stream) {
  (void)in_sizes; (void)n_in; (void)out_size; (void)ws_size;

  const float* x = (const float*)d_in[0];
  const float *wc[6], *cg[6], *cbv[6], *cm[6], *cv[6];
  for (int i = 0; i < 6; ++i) {
    wc[i]  = (const float*)d_in[1 + 5 * i];
    cg[i]  = (const float*)d_in[2 + 5 * i];
    cbv[i] = (const float*)d_in[3 + 5 * i];
    cm[i]  = (const float*)d_in[4 + 5 * i];
    cv[i]  = (const float*)d_in[5 + 5 * i];
  }
  const float *wf[2], *fg[2], *fb[2], *fm[2], *fv[2];
  for (int i = 0; i < 2; ++i) {
    wf[i] = (const float*)d_in[31 + 5 * i];
    fg[i] = (const float*)d_in[32 + 5 * i];
    fb[i] = (const float*)d_in[33 + 5 * i];
    fm[i] = (const float*)d_in[34 + 5 * i];
    fv[i] = (const float*)d_in[35 + 5 * i];
  }
  const float* wf2  = (const float*)d_in[41];
  const float* tn_w = (const float*)d_in[42];
  const float* tn_b = (const float*)d_in[43];
  const float* tn_m = (const float*)d_in[44];
  const float* tn_v = (const float*)d_in[45];
  float* out = (float*)d_out;

  char* ws = (char*)d_ws;
  size_t off = 0;
  auto alloc = [&](size_t bytes) { size_t o = off; off += (bytes + 255) & ~(size_t)255; return o; };
  const size_t oWC0 = alloc(64 * 27);
  const size_t oWC1 = alloc(64 * 576);
  const size_t oWC2 = alloc(128 * 576);
  const size_t oWC3 = alloc(128 * 1152);
  const size_t oWC4 = alloc(256 * 1152);
  const size_t oWC5 = alloc(256 * 2304);
  const size_t oWF0 = alloc(512 * 256);
  const size_t oWF1 = alloc(512 * 512);
  const size_t oWF2 = alloc(64 * 512);
  const size_t oXB  = alloc((size_t)512 * 32 * 32 * 3);
  const size_t oBUFA = alloc((size_t)512 * 30 * 30 * 64);
  const size_t oBUFB = alloc((size_t)512 * 28 * 28 * 64);
  int8_t* bufA = (int8_t*)(ws + oBUFA);
  int8_t* bufB = (int8_t*)(ws + oBUFB);
  int8_t* xb   = (int8_t*)(ws + oXB);

  auto packC = [&](const float* w, size_t o, int Co, int Ci) {
    int tot = Co * Ci * 9;
    k_pack_conv<<<ceil_div(tot, 256), 256, 0, stream>>>(w, (int8_t*)(ws + o), Co, Ci, 3, 3);
  };
  packC(wc[0], oWC0, 64, 3);
  packC(wc[1], oWC1, 64, 64);
  packC(wc[2], oWC2, 128, 64);
  packC(wc[3], oWC3, 128, 128);
  packC(wc[4], oWC4, 256, 128);
  packC(wc[5], oWC5, 256, 256);
  k_pack_fc<<<ceil_div(512 * 256, 256), 256, 0, stream>>>(wf[0], (int8_t*)(ws + oWF0), 512, 256, 512);
  k_pack_fc<<<ceil_div(512 * 512, 256), 256, 0, stream>>>(wf[1], (int8_t*)(ws + oWF1), 512, 512, 512);
  k_pack_fc<<<ceil_div(64 * 512, 256), 256, 0, stream>>>(wf2, (int8_t*)(ws + oWF2), 10, 512, 64);

  k_binarize_input<<<ceil_div(512 * 32 * 32 * 3, 256), 256, 0, stream>>>(x, xb);
  k_conv0<<<512 * 30 * 30 / 256, 256, 0, stream>>>(
      xb, (int8_t*)(ws + oWC0), cg[0], cbv[0], cm[0], cv[0], bufA);

  auto gemm = [&](const int8_t* Ain, size_t wOff, int8_t* o8, float* of,
                  const float* g, const float* b, const float* m, const float* v,
                  int IH, int IW, int Cin, int OH, int OW, int Cout, int CoutTrue,
                  int KH, int KW, float eps, int mode) {
    int M = 512 * OH * OW;
    dim3 grid(M / 256, Cout / 64);
    bnn_gemm_wmma<<<grid, 256, 0, stream>>>(
        Ain, (const int8_t*)(ws + wOff), o8, of, g, b, m, v,
        IH, IW, Cin, OH, OW, Cout, CoutTrue, KH, KW, eps, mode);
  };

  gemm(bufA, oWC1, bufB, nullptr, cg[1], cbv[1], cm[1], cv[1],
       30, 30, 64, 28, 28, 64, 64, 3, 3, 1e-4f, 0);
  k_maxpool2x2<<<512 * 14 * 14 * 64 / 256, 256, 0, stream>>>(bufB, bufA, 28, 28, 64, 512 * 14 * 14 * 64);
  gemm(bufA, oWC2, bufB, nullptr, cg[2], cbv[2], cm[2], cv[2],
       14, 14, 64, 12, 12, 128, 128, 3, 3, 1e-4f, 0);
  gemm(bufB, oWC3, bufA, nullptr, cg[3], cbv[3], cm[3], cv[3],
       12, 12, 128, 10, 10, 128, 128, 3, 3, 1e-4f, 0);
  k_maxpool2x2<<<512 * 5 * 5 * 128 / 256, 256, 0, stream>>>(bufA, bufB, 10, 10, 128, 512 * 5 * 5 * 128);
  gemm(bufB, oWC4, bufA, nullptr, cg[4], cbv[4], cm[4], cv[4],
       5, 5, 128, 3, 3, 256, 256, 3, 3, 1e-4f, 0);
  gemm(bufA, oWC5, bufB, nullptr, cg[5], cbv[5], cm[5], cv[5],
       3, 3, 256, 1, 1, 256, 256, 3, 3, 1e-4f, 0);
  gemm(bufB, oWF0, bufA, nullptr, fg[0], fb[0], fm[0], fv[0],
       1, 1, 256, 1, 1, 512, 512, 1, 1, 1e-5f, 0);
  gemm(bufA, oWF1, bufB, nullptr, fg[1], fb[1], fm[1], fv[1],
       1, 1, 512, 1, 1, 512, 512, 1, 1, 1e-5f, 0);
  gemm(bufB, oWF2, nullptr, out, tn_w, tn_b, tn_m, tn_v,
       1, 1, 512, 1, 1, 64, 10, 1, 1, 1e-4f, 1);
}
